// DeformConv2dPack_78546361909940
// MI455X (gfx1250) — matrix operand order repaired and checked by reading, not compiled, not run on hardware
//
#include <hip/hip_runtime.h>
#include <cstdint>
#include <cstddef>

typedef __attribute__((ext_vector_type(16))) __bf16        v16bf;
typedef __attribute__((ext_vector_type(8)))  float         v8f;
typedef __attribute__((ext_vector_type(4)))  float         f32x4;
typedef __attribute__((ext_vector_type(4)))  unsigned int  u32x4;

#define CIN    128
#define COUT   128
#define HW     64
#define KTAPS  9
#define KTOT   (CIN * KTAPS)   // 1152
#define NCHUNK (KTOT / 32)     // 36 K-steps of 32
#define OMC    27              // offset(18) + mask(9)

// workspace layout (bytes)
#define WS_WDEF   0                         // bf16 [36][128][2][16] = 294912 B
#define WS_WOFF   294912                    // bf16 [36][32][2][16]  =  73728 B
#define WS_OM     368640                    // f32  [B*64*64][32]    = 4194304 B
#define WS_XT     4562944                   // f32  NHWC x           = 16777216 B

__device__ __forceinline__ unsigned short f2bf(float f) {
  union { float f; unsigned int u; } c; c.f = f;
  unsigned int u = c.u;
  u += 0x7FFFu + ((u >> 16) & 1u);        // round-to-nearest-even
  return (unsigned short)(u >> 16);
}

__device__ __forceinline__ unsigned int pk_bf16(float a, float b) {
#if __has_builtin(__builtin_amdgcn_cvt_pk_bf16_f32)
  auto r = __builtin_amdgcn_cvt_pk_bf16_f32(a, b);
  unsigned int u;
  __builtin_memcpy(&u, &r, 4);
  return u;
#elif defined(__gfx1250__)
  unsigned int u;
  asm("v_cvt_pk_bf16_f32 %0, %1, %2" : "=v"(u) : "v"(a), "v"(b));
  return u;
#else
  return (unsigned int)f2bf(a) | ((unsigned int)f2bf(b) << 16);
#endif
}

union Frag { v16bf v; u32x4 q[2]; };

// ---------------- NCHW -> NHWC transpose of x ------------------------------
__global__ void __launch_bounds__(256) transpose_x_kernel(
    const float* __restrict__ x, float* __restrict__ xt) {
  int t = threadIdx.x;
  int row = blockIdx.x;                 // b*64 + y
  int b = row >> 6, y = row & 63;
  int xx = t & 63, cg = t >> 6;         // 4 groups of 32 channels
  const float* src = x + ((size_t)(b * CIN) << 12) + (y << 6) + xx;
  float* dst = xt + ((size_t)row * 64 + xx) * CIN;
#pragma unroll
  for (int i0 = 0; i0 < 32; i0 += 4) {
    int c = cg * 32 + i0;
    f32x4 v;
    v.x = src[(size_t)(c + 0) << 12];
    v.y = src[(size_t)(c + 1) << 12];
    v.z = src[(size_t)(c + 2) << 12];
    v.w = src[(size_t)(c + 3) << 12];
    *(f32x4*)&dst[c] = v;
  }
}

// ---------------- weight packing into WMMA B-fragment order ----------------
__global__ void pack_wdef_kernel(const float* __restrict__ w,
                                 unsigned short* __restrict__ o) {
  int idx = blockIdx.x * blockDim.x + threadIdx.x;
  if (idx >= NCHUNK * COUT * 32) return;
  int j = idx & 15, g = (idx >> 4) & 1, n = (idx >> 5) & (COUT - 1),
      chunk = idx >> 12;
  int c = g * 16 + j;                 // K within chunk
  int kk = chunk * 32 + c;            // kk = k*CIN + cin
  int k = kk >> 7, cin = kk & (CIN - 1);
  o[idx] = f2bf(w[(n * CIN + cin) * KTAPS + k]);
}

__global__ void pack_woff_kernel(const float* __restrict__ w,
                                 unsigned short* __restrict__ o) {
  int idx = blockIdx.x * blockDim.x + threadIdx.x;
  if (idx >= NCHUNK * 32 * 32) return;
  int j = idx & 15, g = (idx >> 4) & 1, n = (idx >> 5) & 31, chunk = idx >> 10;
  int c = g * 16 + j;
  int kk = chunk * 32 + c;
  int k = kk >> 7, cin = kk & (CIN - 1);
  o[idx] = f2bf(n < OMC ? w[(n * CIN + cin) * KTAPS + k] : 0.f);
}

// ---------------- offset conv (implicit GEMM, M=64 N=32 K=1152) ------------
__global__ void __launch_bounds__(256) conv_offset_wmma(
    const float* __restrict__ xt, const unsigned short* __restrict__ wp,
    const float* __restrict__ b_off, float* __restrict__ om) {
  __shared__ unsigned short A_lds[2][64 * 32];
  int t = threadIdx.x;
  int b = blockIdx.x >> 6, ho = blockIdx.x & 63;
  int wid = t >> 5, lane = t & 31;
  int mt = wid & 3, nt = wid >> 2;          // 4 M-tiles x 2 N-tiles
  int m = lane & 15, gh = lane >> 4;
  const float* xbT = xt + (size_t)b * (64 * 64 * CIN);
  int p = t >> 2, q = t & 3;

  f32x4 sa, sb;         // staged raw samples (8 channels) for pending chunk
  float wv = 0.f;

  auto gather = [&](int chunk) {        // 2 contiguous b128 loads (NHWC)
    int k = chunk >> 2, cinb = (chunk & 3) * 32;
    int y  = ho - 1 + k / 3;
    int xx = p  - 1 + k % 3;
    wv = (((unsigned)y < 64u) & ((unsigned)xx < 64u)) ? 1.f : 0.f;
    int yc = min(max(y, 0), 63), xc = min(max(xx, 0), 63);
    const float* s = xbT + (((yc << 6) + xc) << 7) + cinb + q * 8;
    sa = *(const f32x4*)s;
    sb = *(const f32x4*)(s + 4);
  };
  auto store = [&](int buf) {           // pack + one b128 LDS store
    f32x4 va = sa * wv, vb = sb * wv;
    u32x4 pk;
    pk.x = pk_bf16(va.x, va.y); pk.y = pk_bf16(va.z, va.w);
    pk.z = pk_bf16(vb.x, vb.y); pk.w = pk_bf16(vb.z, vb.w);
    *(u32x4*)&A_lds[buf][(p * 2 + (q >> 1)) * 16 + (q & 1) * 8] = pk;
  };

  v8f acc = {};
  gather(0); store(0);
  for (int chunk = 0; chunk < NCHUNK; ++chunk) {
    __syncthreads();
    int buf = chunk & 1;
    Frag a, bm;
    const u32x4* ap = (const u32x4*)&A_lds[buf][((mt * 16 + m) * 2 + gh) * 16];
    a.q[0] = ap[0]; a.q[1] = ap[1];
    int n = nt * 16 + m;
    const u32x4* bp = (const u32x4*)&wp[((chunk * 32 + n) * 2 + gh) * 16];
    bm.q[0] = bp[0]; bm.q[1] = bp[1];
    if (chunk + 1 < NCHUNK) {
      __builtin_prefetch(&wp[(size_t)(chunk + 1) * 1024 + (t & 63) * 16], 0, 3);
      gather(chunk + 1);
    }
    acc = __builtin_amdgcn_wmma_f32_16x16x32_bf16(
        false, a.v, false, bm.v, (short)0, acc, false, false);
    if (chunk + 1 < NCHUNK) store(buf ^ 1);
  }
  // epilogue: bias, sigmoid on mask channels, store per-position [32] row
  int n = nt * 16 + m;
  if (n < OMC) {
    float bias = b_off[n];
#pragma unroll
    for (int r = 0; r < 8; ++r) {
      int wo = mt * 16 + r + 8 * gh;
      float v = acc[r] + bias;
      if (n >= 18) v = __builtin_amdgcn_rcpf(1.f + __expf(-v));
      om[((((size_t)(b * 64 + ho) << 6) + wo) << 5) + n] = v;
    }
  }
}

// ---------------- deformable conv (implicit GEMM, M=64 N=128 K=1152) -------
__global__ void __launch_bounds__(256) deform_conv_wmma(
    const float* __restrict__ xt, const unsigned short* __restrict__ wp,
    const float* __restrict__ om, const float* __restrict__ b_def,
    float* __restrict__ out) {
  __shared__ unsigned short A_lds[2][64 * 32];
  __shared__ int   s_y0[KTAPS][64], s_x0[KTAPS][64];
  __shared__ float s_wy[KTAPS][64], s_wx[KTAPS][64], s_m[KTAPS][64];
  __shared__ float C_lds[COUT * 65];        // pad 65 -> no bank conflicts
  int t = threadIdx.x;
  int b = blockIdx.x >> 6, ho = blockIdx.x & 63;
  int wid = t >> 5, lane = t & 31;
  int mw = wid & 1, nw = wid >> 1;          // 2(M) x 4(N) wave grid
  int m = lane & 15, gh = lane >> 4;
  const float* xbT = xt + (size_t)b * (64 * 64 * CIN);
  int p = t >> 2, q = t & 3;

  // bilinear metadata for all 9 taps, once
  for (int idx = t; idx < KTAPS * 64; idx += 256) {
    int pp = idx & 63, k = idx >> 6;
    const float* r = &om[((size_t)((b * 64 + ho) << 6) + pp) << 5];
    float py = (float)(ho + k / 3 - 1) + r[k];
    float px = (float)(pp + k % 3 - 1) + r[9 + k];
    float fy = floorf(py), fx = floorf(px);
    s_y0[k][pp] = (int)fy; s_x0[k][pp] = (int)fx;
    s_wy[k][pp] = py - fy; s_wx[k][pp] = px - fx;
    s_m[k][pp]  = r[18 + k];
  }

  f32x4 s00a, s00b, s01a, s01b, s10a, s10b, s11a, s11b;  // pending samples
  float w00 = 0.f, w01 = 0.f, w10 = 0.f, w11 = 0.f;

  // phase 1: 8 contiguous b128 gathers (NHWC); validity folded into weights
  auto gather = [&](int chunk) {
    int k = chunk >> 2, cc = chunk & 3;
    int y0 = s_y0[k][p], x0 = s_x0[k][p];
    float wy = s_wy[k][p], wx = s_wx[k][p], mm = s_m[k][p];
    int y1 = y0 + 1, x1 = x0 + 1;
    float fy0 = ((unsigned)y0 < 64u) ? 1.f : 0.f;
    float fy1 = ((unsigned)y1 < 64u) ? 1.f : 0.f;
    float fx0 = ((unsigned)x0 < 64u) ? 1.f : 0.f;
    float fx1 = ((unsigned)x1 < 64u) ? 1.f : 0.f;
    w00 = (1.f - wy) * (1.f - wx) * mm * fy0 * fx0;
    w01 = (1.f - wy) * wx         * mm * fy0 * fx1;
    w10 = wy * (1.f - wx)         * mm * fy1 * fx0;
    w11 = wy * wx                 * mm * fy1 * fx1;
    int yi0 = min(max(y0, 0), 63), yi1 = min(max(y1, 0), 63);
    int xi0 = min(max(x0, 0), 63), xi1 = min(max(x1, 0), 63);
    int cb = cc * 32 + q * 8;
    const float* p00 = xbT + ((((yi0 << 6) + xi0) << 7) + cb);
    const float* p01 = xbT + ((((yi0 << 6) + xi1) << 7) + cb);
    const float* p10 = xbT + ((((yi1 << 6) + xi0) << 7) + cb);
    const float* p11 = xbT + ((((yi1 << 6) + xi1) << 7) + cb);
    s00a = *(const f32x4*)p00; s00b = *(const f32x4*)(p00 + 4);
    s01a = *(const f32x4*)p01; s01b = *(const f32x4*)(p01 + 4);
    s10a = *(const f32x4*)p10; s10b = *(const f32x4*)(p10 + 4);
    s11a = *(const f32x4*)p11; s11b = *(const f32x4*)(p11 + 4);
  };
  // phase 2: vector bilinear combine, pack, one b128 LDS store
  auto store = [&](int buf) {
    f32x4 va = s00a * w00 + s01a * w01 + s10a * w10 + s11a * w11;
    f32x4 vb = s00b * w00 + s01b * w01 + s10b * w10 + s11b * w11;
    u32x4 pk;
    pk.x = pk_bf16(va.x, va.y); pk.y = pk_bf16(va.z, va.w);
    pk.z = pk_bf16(vb.x, vb.y); pk.w = pk_bf16(vb.z, vb.w);
    *(u32x4*)&A_lds[buf][(p * 2 + (q >> 1)) * 16 + (q & 1) * 8] = pk;
  };

  v8f acc[2][2] = {};
  __syncthreads();          // metadata visible
  gather(0); store(0);
  for (int chunk = 0; chunk < NCHUNK; ++chunk) {
    __syncthreads();        // buf[chunk&1] staged; other buffer free
    int buf = chunk & 1;
    // fragment loads first (counted wait leaves next gathers in flight)
    Frag a[2], bf[2];
#pragma unroll
    for (int mi = 0; mi < 2; ++mi) {
      int mtile = mw * 2 + mi;
      const u32x4* ap = (const u32x4*)&A_lds[buf][((mtile * 16 + m) * 2 + gh) * 16];
      a[mi].q[0] = ap[0]; a[mi].q[1] = ap[1];
    }
#pragma unroll
    for (int ni = 0; ni < 2; ++ni) {
      int n = (nw * 2 + ni) * 16 + m;
      const u32x4* bp = (const u32x4*)&wp[((size_t)(chunk * COUT + n) * 2 + gh) * 16];
      bf[ni].q[0] = bp[0]; bf[ni].q[1] = bp[1];
    }
    if (chunk + 1 < NCHUNK) {
      __builtin_prefetch(&wp[(size_t)(chunk + 1) * 4096 + t * 16], 0, 3);
      gather(chunk + 1);    // 8 b128 loads overlap the WMMAs below
    }
#pragma unroll
    for (int mi = 0; mi < 2; ++mi)
#pragma unroll
      for (int ni = 0; ni < 2; ++ni)
        acc[mi][ni] = __builtin_amdgcn_wmma_f32_16x16x32_bf16(
            false, a[mi].v, false, bf[ni].v, (short)0, acc[mi][ni], false, false);
    if (chunk + 1 < NCHUNK) store(buf ^ 1);
  }
  // epilogue: transpose through LDS, then vectorized coalesced stores
#pragma unroll
  for (int mi = 0; mi < 2; ++mi)
#pragma unroll
    for (int ni = 0; ni < 2; ++ni) {
      int cout = (nw * 2 + ni) * 16 + m;
#pragma unroll
      for (int r = 0; r < 8; ++r) {
        int wo = (mw * 2 + mi) * 16 + r + 8 * gh;
        C_lds[cout * 65 + wo] = acc[mi][ni][r];
      }
    }
  __syncthreads();
  {
    int cout = t >> 1, wob = (t & 1) * 32;
    float bias = b_def[cout];
    float* op = &out[((size_t)(((b * COUT + cout) << 6) + ho)) << 6];
    const float* cp = &C_lds[cout * 65 + wob];
#pragma unroll
    for (int i = 0; i < 8; ++i) {
      f32x4 v;
      v.x = cp[i * 4 + 0] + bias; v.y = cp[i * 4 + 1] + bias;
      v.z = cp[i * 4 + 2] + bias; v.w = cp[i * 4 + 3] + bias;
      *(f32x4*)&op[wob + i * 4] = v;
    }
  }
}

extern "C" void kernel_launch(void* const* d_in, const int* in_sizes, int n_in,
                              void* d_out, int out_size, void* d_ws, size_t ws_size,
                              hipStream_t stream) {
  (void)in_sizes; (void)n_in; (void)out_size; (void)ws_size;
  const float* x     = (const float*)d_in[0];
  const float* w_off = (const float*)d_in[1];
  const float* b_off = (const float*)d_in[2];
  const float* w_def = (const float*)d_in[3];
  const float* b_def = (const float*)d_in[4];
  float* out = (float*)d_out;
  char*  ws  = (char*)d_ws;
  unsigned short* wdefp   = (unsigned short*)(ws + WS_WDEF);
  unsigned short* woffp   = (unsigned short*)(ws + WS_WOFF);
  float*          offmask = (float*)(ws + WS_OM);
  float*          xT      = (float*)(ws + WS_XT);

  pack_wdef_kernel<<<(NCHUNK * COUT * 32 + 255) / 256, 256, 0, stream>>>(w_def, wdefp);
  pack_woff_kernel<<<(NCHUNK * 32 * 32 + 255) / 256, 256, 0, stream>>>(w_off, woffp);
  transpose_x_kernel<<<8 * 64, 256, 0, stream>>>(x, xT);
  conv_offset_wmma<<<8 * 64, 256, 0, stream>>>(xT, woffp, b_off, offmask);
  deform_conv_wmma<<<8 * 64, 256, 0, stream>>>(xT, wdefp, offmask, b_def, out);
}
